// MS_SSA_Conv_58248346468848
// MI455X (gfx1250) — matrix operand order repaired and checked by reading, not compiled, not run on hardware
//
#include <hip/hip_runtime.h>

// ---------------------------------------------------------------------------
// MS-SSA (spiking self-attention) for MI455X / gfx1250, wave32 + WMMA f16.
// All GEMMs are "TN" (both operands K-contiguous). Weight slabs are staged in
// LDS in fragment-swizzled order; k/v slabs in k_attn are staged by the
// Tensor Data Mover (TDM) with LDS row padding; activation fragments are
// double-buffered in registers to hide loadcnt.
// ---------------------------------------------------------------------------

typedef _Float16 half_t;
typedef __attribute__((ext_vector_type(8)))  _Float16 v8h;
typedef __attribute__((ext_vector_type(16))) _Float16 v16h;
typedef __attribute__((ext_vector_type(8)))  float    v8f;
typedef __attribute__((ext_vector_type(4)))  unsigned int v4u;
typedef __attribute__((ext_vector_type(4)))  int v4i;
typedef __attribute__((ext_vector_type(8)))  int v8i;

#define TT    4
#define BB    8
#define CC    512
#define NN    1024
#define HEADS 8
#define HD    64
#define CSN   2048   // CHUNK * N

#if defined(__has_builtin)
#  if __has_builtin(__builtin_amdgcn_tensor_load_to_lds)
#    define HAVE_TDM 1
#  endif
#endif

// --- fragment loader: storage is [row][k] row-major, k-contiguous ----------
// ISA 16-bit A/B layout (16x32): lane l<16 holds row l, K={k0..k0+7, k0+16..k0+23};
// lane l+16 holds row l, K shifted by 8. Two 16B loads per lane.
__device__ __forceinline__ v16h load_frag(const half_t* base, int row0, int ld, int k0) {
  const int lane = threadIdx.x & 31;
  const int r = lane & 15, ks = lane >> 4;
  const half_t* p = base + (size_t)(row0 + r) * ld + (k0 + ks * 8);
  union { v16h w; v8h h[2]; } u;
  u.h[0] = *(const v8h*)(p);
  u.h[1] = *(const v8h*)(p + 16);
  return u.w;
}

// --- fragment loader from LDS, fragment-swizzled layout --------------------
// frag = kk*4 + i; lane's 16 halves are contiguous at frag*512 + lane*16.
__device__ __forceinline__ v16h load_frag_sw(const half_t* sm, int frag) {
  const int lane = threadIdx.x & 31;
  const half_t* p = sm + frag * 512 + lane * 16;
  union { v16h w; v8h h[2]; } u;
  u.h[0] = *(const v8h*)(p);
  u.h[1] = *(const v8h*)(p + 8);
  return u.w;
}

// --- cooperative fill of a 64-row x 512-k weight slab into swizzled LDS ----
__device__ __forceinline__ void fill_weight_slab(half_t* sm,
                                                 const half_t* __restrict__ wsrc,
                                                 int base_row) {
  for (int m = threadIdx.x; m < 4096; m += 256) {
    int frag = m >> 6;
    int kk = frag >> 2, i = frag & 3;
    int c = m & 63;
    int lane = c >> 1, hs = c & 1;
    int r = lane & 15, ks = lane >> 4;
    int row = base_row + i * 16 + r;
    int k = kk * 32 + hs * 16 + ks * 8;
    *(v8h*)(sm + (size_t)m * 8) = *(const v8h*)(wsrc + (size_t)row * CC + k);
  }
}

__device__ __forceinline__ v8f wmma_f16(v16h a, v16h b, v8f c) {
  return __builtin_amdgcn_wmma_f32_16x16x32_f16(false, a, false, b,
                                                (short)0, c, false, false);
}

#ifdef HAVE_TDM
// --- TDM: 2D tile (rows x tile0 elements of 2B) global -> LDS with padding -
// D# per CDNA5 ISA: group0 {count=1, lds_addr,
// global_addr, type=2}; group1 {data_size=2B, pad_enable, pad_interval code,
// pad_amount code, tensor_dim0/1, tile_dim0/1, tensor_dim0_stride}.
__device__ __forceinline__ void tdm_load_tile(unsigned lds_off_bytes, const void* gsrc,
                                              unsigned dim0, unsigned tile0, unsigned rows,
                                              unsigned stride0, unsigned pi_code,
                                              unsigned pa_code) {
  unsigned long long ga = (unsigned long long)gsrc;
  v4u g0 = { 1u,                                    // count=1 (valid descriptor)
             lds_off_bytes,                         // lds_addr
             (unsigned)ga,                          // global_addr[31:0]
             (unsigned)((ga >> 32) & 0x01FFFFFFu) | 0x80000000u };  // addr[56:32] | type=2
  v8i g1 = { (int)((1u << 16) | (1u << 20) | (pi_code << 22) | (pa_code << 25)),
             (int)((dim0 & 0xffffu) << 16),                          // tensor_dim0 lo16
             (int)((dim0 >> 16) | ((rows & 0xffffu) << 16)),         // dim0 hi | dim1 lo
             (int)((rows >> 16) | ((tile0 & 0xffffu) << 16)),        // dim1 hi | tile_dim0
             (int)(rows & 0xffffu),                                  // tile_dim1 (tile_dim2=0)
             (int)stride0,                                           // dim0_stride[31:0]
             0, 0 };
  v4i z4 = { 0, 0, 0, 0 };
#if __clang_major__ >= 23
  v8i z8 = { 0, 0, 0, 0, 0, 0, 0, 0 };
  __builtin_amdgcn_tensor_load_to_lds(g0, g1, z4, z4, z8, 0);
#else
  __builtin_amdgcn_tensor_load_to_lds(g0, g1, z4, z4, 0);
#endif
}

__device__ __forceinline__ void tdm_wait() {
#if __has_builtin(__builtin_amdgcn_s_wait_tensorcnt)
  __builtin_amdgcn_s_wait_tensorcnt(0);
#else
  asm volatile("s_wait_tensorcnt 0x0" ::: "memory");
#endif
}
#endif  // HAVE_TDM

// ---------------------------------------------------------------------------
// K0: fold BN scale into weights (fp32 -> f16), build fused bias.
// ---------------------------------------------------------------------------
__global__ void k_fold(const float* __restrict__ w, const float* __restrict__ gamma,
                       const float* __restrict__ beta, const float* __restrict__ mean,
                       const float* __restrict__ var, const float* __restrict__ convb,
                       half_t* __restrict__ wout, float* __restrict__ bout) {
  int gid = blockIdx.x * 256 + threadIdx.x;
  int o = gid >> 9, c = gid & 511;
  float scale = gamma[o] * rsqrtf(var[o] + 1e-5f);
  wout[gid] = (half_t)(w[gid] * scale);
  if (c == 0) {
    float cb = convb ? convb[o] * scale : 0.0f;
    bout[o] = cb + beta[o] - mean[o] * scale;
  }
}

// ---------------------------------------------------------------------------
// K1: shortcut LIF over x, with 16x16 LDS transpose [T,B,C,N] -> xs[T,B,N,C]
// ---------------------------------------------------------------------------
__global__ void k_lif_x(const float* __restrict__ x, half_t* __restrict__ xs) {
  __shared__ half_t sm[TT][16][17];
  const int b  = blockIdx.z;
  const int c0 = blockIdx.y * 16;
  const int n0 = blockIdx.x * 16;
  const int tx = threadIdx.x & 15, ty = threadIdx.x >> 4;
  float v = 0.0f;
  #pragma unroll
  for (int t = 0; t < TT; ++t) {
    float xv = x[((size_t)(t * BB + b) * CC + c0 + ty) * NN + n0 + tx];
    v = 0.5f * (v + xv);                 // v += (x - v)/tau, tau = 2
    float s = (v >= 1.0f) ? 1.0f : 0.0f; // H(v - 1)
    v = (s != 0.0f) ? 0.0f : v;          // hard reset
    sm[t][ty][tx] = (half_t)s;
  }
  __syncthreads();
  #pragma unroll
  for (int t = 0; t < TT; ++t)
    xs[((size_t)(t * BB + b) * NN + n0 + ty) * CC + c0 + tx] = sm[t][tx][ty];
}

// ---------------------------------------------------------------------------
// K2: fused q/k/v conv1x1 GEMM + folded BN + LIF (v_th = 1).
//   Weight slab (64x512, shared by all 8 waves) in swizzled LDS; the four
//   a-frags of a k-step are loaded as distinct live values before the four
//   WMMAs (disjoint regs -> no WMMA WAR v_nops); xs b-frag double-buffered.
// ---------------------------------------------------------------------------
__global__ void k_qkv(const half_t* __restrict__ xs, const half_t* __restrict__ wqkv,
                      const float* __restrict__ bqkv, half_t* __restrict__ qbuf,
                      half_t* __restrict__ ktbuf, half_t* __restrict__ vtbuf) {
  __shared__ half_t wsm[64 * 512];                 // 64 KB swizzled weight slab
  const int b = blockIdx.z, og = blockIdx.y, nb = blockIdx.x;
  const int w = threadIdx.x >> 5, lane = threadIdx.x & 31;
  const int ks = lane >> 4, col = lane & 15;
  const int n0 = nb * 128 + w * 16;
  const int mat = og >> 3, head = og & 7;

  fill_weight_slab(wsm, wqkv, og * 64);
  __syncthreads();

  float vst[4][8] = {};
  for (int t = 0; t < TT; ++t) {
    v8f acc[4] = {};
    const half_t* bbase = xs + ((size_t)(t * BB + b) * NN + n0) * CC;
    v16h bf = load_frag(bbase, 0, CC, 0);
    #pragma unroll 4
    for (int kk = 0; kk < 16; ++kk) {
      v16h bcur = bf;
      if (kk + 1 < 16) bf = load_frag(bbase, 0, CC, (kk + 1) * 32);
      v16h a0 = load_frag_sw(wsm, kk * 4 + 0);
      v16h a1 = load_frag_sw(wsm, kk * 4 + 1);
      v16h a2 = load_frag_sw(wsm, kk * 4 + 2);
      v16h a3 = load_frag_sw(wsm, kk * 4 + 3);
      acc[0] = wmma_f16(a0, bcur, acc[0]);
      acc[1] = wmma_f16(a1, bcur, acc[1]);
      acc[2] = wmma_f16(a2, bcur, acc[2]);
      acc[3] = wmma_f16(a3, bcur, acc[3]);
    }
    const int chunk = t >> 1;
    const int pos = (t & 1) * NN + n0 + col;
    const size_t hb = ((size_t)(chunk * BB + b) * HEADS + head);
    #pragma unroll
    for (int i = 0; i < 4; ++i) {
      #pragma unroll
      for (int r = 0; r < 8; ++r) {
        float u = acc[i][r] + bqkv[og * 64 + i * 16 + r + 8 * ks];
        float m = 0.5f * (vst[i][r] + u);
        float s = (m >= 1.0f) ? 1.0f : 0.0f;
        vst[i][r] = (s != 0.0f) ? 0.0f : m;
        int d = i * 16 + r + 8 * ks;
        half_t sh = (half_t)s;
        if (mat == 0)      qbuf [(hb * CSN + pos) * HD + d] = sh;
        else if (mat == 1) ktbuf[(hb * HD + d) * CSN + pos] = sh;
        else               vtbuf[(hb * HD + d) * CSN + pos] = sh;
      }
    }
  }
}

// ---------------------------------------------------------------------------
// K3: one block per (chunk,b,head).
//   Phase 1: kT/vT slabs (128 positions) staged in LDS via TDM (padded rows,
//            272B pitch), shared by all 8 waves; attnT -> LDS transposed.
//   Phase 2: out[pos][e] = sum_d q[pos][d]*attnT[e][d], b-frags cached in regs.
// ---------------------------------------------------------------------------
#define KV_LD 136                       // 128 halves + 8 pad (16B) per row
__global__ void k_attn(const half_t* __restrict__ qbuf, const half_t* __restrict__ ktbuf,
                       const half_t* __restrict__ vtbuf, float* __restrict__ outb) {
  __shared__ half_t smem[2 * 64 * KV_LD + 64 * 72];   // 44,032 B total
  half_t* kslab = smem;                               // LDS byte off 0
  half_t* vslab = smem + 64 * KV_LD;                  // LDS byte off 17408
  half_t* at    = smem + 2 * 64 * KV_LD;              // attnT, pad 8/row

  const int id = blockIdx.x;
  const int h = id & 7, b = (id >> 3) & 7, chunk = id >> 6;
  const int w = threadIdx.x >> 5, lane = threadIdx.x & 31;
  const int ks = lane >> 4, col = lane & 15;
  const size_t hb = ((size_t)(chunk * BB + b) * HEADS + h);
  const half_t* kb = ktbuf + hb * HD * CSN;
  const half_t* vb = vtbuf + hb * HD * CSN;

  // ---- phase 1: 16 slabs x 4 k-steps over K=2048; wave -> 1 d-tile x 2 e-tiles
  const int dt = w >> 1, ep = (w & 1) * 2;
  v8f a0 = {}, a1 = {};
  for (int s = 0; s < 16; ++s) {
#ifdef HAVE_TDM
    if (threadIdx.x == 0) {
      // tile 128x64 of 2B elems; pad_interval code 5 (=64 DW = one 256B row),
      // pad_amount code 3 (=4 DW = 16B) -> 272B LDS row pitch
      tdm_load_tile(0u,     kb + (size_t)s * 128, CSN, 128, 64, CSN, 5u, 3u);
      tdm_load_tile(17408u, vb + (size_t)s * 128, CSN, 128, 64, CSN, 5u, 3u);
      tdm_wait();
    }
#else
    for (int m = threadIdx.x; m < 1024; m += 256) {   // 64 rows x 16 chunks of 16B
      int row = m >> 4, ch = m & 15;
      *(v8h*)(kslab + row * KV_LD + ch * 8) =
          *(const v8h*)(kb + (size_t)row * CSN + s * 128 + ch * 8);
      *(v8h*)(vslab + row * KV_LD + ch * 8) =
          *(const v8h*)(vb + (size_t)row * CSN + s * 128 + ch * 8);
    }
#endif
    __syncthreads();
    #pragma unroll
    for (int kkl = 0; kkl < 4; ++kkl) {
      v16h ac = load_frag(kslab, dt * 16, KV_LD, kkl * 32);
      v16h c0 = load_frag(vslab, ep * 16, KV_LD, kkl * 32);
      v16h c1 = load_frag(vslab, (ep + 1) * 16, KV_LD, kkl * 32);
      a0 = wmma_f16(ac, c0, a0);
      a1 = wmma_f16(ac, c1, a1);
    }
    __syncthreads();
  }
  const float sc = 1.0f / 32.0f;                      // scale = 1/H
  #pragma unroll
  for (int r = 0; r < 8; ++r) {
    int d = dt * 16 + r + 8 * ks;
    at[(ep * 16 + col) * 72 + d]       = (half_t)(a0[r] * sc);
    at[((ep + 1) * 16 + col) * 72 + d] = (half_t)(a1[r] * sc);
  }
  __syncthreads();

  // ---- phase 2: cache all attnT b-frags (ds_load_b128), sweep 128 pos-tiles
  v16h bf[4][2];
  #pragma unroll
  for (int e = 0; e < 4; ++e)
    #pragma unroll
    for (int kk = 0; kk < 2; ++kk)
      bf[e][kk] = load_frag(at, e * 16, 72, kk * 32);

  const half_t* qb = qbuf + hb * CSN * HD;
  for (int pt = w; pt < 128; pt += 8) {
    const int pos0 = pt * 16;
    v16h af0 = load_frag(qb, pos0, HD, 0);
    v16h af1 = load_frag(qb, pos0, HD, 32);
    v8f acc[4] = {};
    #pragma unroll
    for (int e = 0; e < 4; ++e) {
      acc[e] = wmma_f16(af0, bf[e][0], acc[e]);
      acc[e] = wmma_f16(af1, bf[e][1], acc[e]);
    }
    const int csi = pos0 >> 10, nbase = pos0 & 1023;
    const int t = chunk * 2 + csi;
    const size_t ob = ((size_t)(t * BB + b) * HEADS + h) * NN;
    #pragma unroll
    for (int e = 0; e < 4; ++e)
      #pragma unroll
      for (int r = 0; r < 8; ++r)
        outb[(ob + nbase + r + 8 * ks) * HD + e * 16 + col] = acc[e][r];
  }
}

// ---------------------------------------------------------------------------
// K4: attn LIF (v_th = 0.5) -> spikes xas[T,B,N,C], c = head*64 + d
// ---------------------------------------------------------------------------
__global__ void k_attn_lif(const float* __restrict__ outb, half_t* __restrict__ xas) {
  const unsigned gid = blockIdx.x * 256 + threadIdx.x;
  const int d = gid & 63;
  const int n = (gid >> 6) & 1023;
  const int h = (gid >> 16) & 7;
  const int b = gid >> 19;
  float v = 0.0f;
  #pragma unroll
  for (int t = 0; t < TT; ++t) {
    float u = outb[(((size_t)(t * BB + b) * HEADS + h) * NN + n) * HD + d];
    v = 0.5f * (v + u);
    float s = (v >= 0.5f) ? 1.0f : 0.0f;
    v = (s != 0.0f) ? 0.0f : v;
    xas[((size_t)(t * BB + b) * NN + n) * CC + h * 64 + d] = (half_t)s;
  }
}

// ---------------------------------------------------------------------------
// K5: proj conv1x1 GEMM + fused bias/BN + residual add -> d_out [T,B,C,H,W]
// ---------------------------------------------------------------------------
__global__ void k_proj(const half_t* __restrict__ xas, const half_t* __restrict__ wproj,
                       const float* __restrict__ bproj, const float* __restrict__ x,
                       float* __restrict__ out) {
  __shared__ half_t wsm[64 * 512];
  const int tb = blockIdx.z, og = blockIdx.y, nb = blockIdx.x;
  const int w = threadIdx.x >> 5, lane = threadIdx.x & 31;
  const int ks = lane >> 4, col = lane & 15;
  const int n0 = nb * 128 + w * 16;

  fill_weight_slab(wsm, wproj, og * 64);
  __syncthreads();

  const half_t* bbase = xas + ((size_t)tb * NN + n0) * CC;
  v8f acc[4] = {};
  v16h bf = load_frag(bbase, 0, CC, 0);
  #pragma unroll 4
  for (int kk = 0; kk < 16; ++kk) {
    v16h bcur = bf;
    if (kk + 1 < 16) bf = load_frag(bbase, 0, CC, (kk + 1) * 32);
    v16h a0 = load_frag_sw(wsm, kk * 4 + 0);
    v16h a1 = load_frag_sw(wsm, kk * 4 + 1);
    v16h a2 = load_frag_sw(wsm, kk * 4 + 2);
    v16h a3 = load_frag_sw(wsm, kk * 4 + 3);
    acc[0] = wmma_f16(a0, bcur, acc[0]);
    acc[1] = wmma_f16(a1, bcur, acc[1]);
    acc[2] = wmma_f16(a2, bcur, acc[2]);
    acc[3] = wmma_f16(a3, bcur, acc[3]);
  }
  const int n = n0 + col;
  #pragma unroll
  for (int i = 0; i < 4; ++i)
    #pragma unroll
    for (int r = 0; r < 8; ++r) {
      int o = og * 64 + i * 16 + r + 8 * ks;
      size_t idx = ((size_t)tb * CC + o) * NN + n;
      out[idx] = acc[i][r] + bproj[o] + x[idx];
    }
}

// ---------------------------------------------------------------------------
// host launch
// ---------------------------------------------------------------------------
extern "C" void kernel_launch(void* const* d_in, const int* in_sizes, int n_in,
                              void* d_out, int out_size, void* d_ws, size_t ws_size,
                              hipStream_t stream) {
  (void)in_sizes; (void)n_in; (void)out_size; (void)ws_size;
  const float* x = (const float*)d_in[0];

  char* ws = (char*)d_ws;
  const size_t SZ_WQKV = (size_t)1536 * 512 * 2;
  const size_t SZ_BQKV = 1536 * 4;
  const size_t SZ_WPRJ = (size_t)512 * 512 * 2;
  const size_t SZ_BPRJ = 512 * 4;
  const size_t SZ_SPK  = (size_t)TT * BB * NN * CC * 2;  // 33.55 MB
  const size_t SZ_OUTB = (size_t)TT * BB * NN * CC * 4;  // 67.1 MB

  size_t off = 0;
  half_t* WQKV = (half_t*)(ws + off); off += SZ_WQKV;
  float*  BQKV = (float*) (ws + off); off += SZ_BQKV;
  half_t* WPRJ = (half_t*)(ws + off); off += SZ_WPRJ;
  float*  BPRJ = (float*) (ws + off); off += SZ_BPRJ;
  size_t off_xs = off;
  half_t* XS   = (half_t*)(ws + off); off += SZ_SPK;
  half_t* QB   = (half_t*)(ws + off); off += SZ_SPK;
  half_t* KT   = (half_t*)(ws + off); off += SZ_SPK;
  half_t* VT   = (half_t*)(ws + off); off += SZ_SPK;
  float*  OUTB = (float*) (ws + off); off += SZ_OUTB;
  half_t* XAS  = (half_t*)(ws + off_xs);                 // alias XS (dead after K2)

  for (int m = 0; m < 3; ++m) {
    k_fold<<<1024, 256, 0, stream>>>(
        (const float*)d_in[1 + 5 * m], (const float*)d_in[2 + 5 * m],
        (const float*)d_in[3 + 5 * m], (const float*)d_in[4 + 5 * m],
        (const float*)d_in[5 + 5 * m], nullptr,
        WQKV + (size_t)m * 512 * 512, BQKV + m * 512);
  }
  k_fold<<<1024, 256, 0, stream>>>((const float*)d_in[16], (const float*)d_in[18],
                                   (const float*)d_in[19], (const float*)d_in[20],
                                   (const float*)d_in[21], (const float*)d_in[17],
                                   WPRJ, BPRJ);

  k_lif_x<<<dim3(64, 32, 8), 256, 0, stream>>>(x, XS);
  k_qkv<<<dim3(8, 24, 8), 256, 0, stream>>>(XS, WQKV, BQKV, QB, KT, VT);
  k_attn<<<dim3(128), 256, 0, stream>>>(QB, KT, VT, OUTB);
  k_attn_lif<<<dim3(16384), 256, 0, stream>>>(OUTB, XAS);
  k_proj<<<dim3(8, 8, 32), 256, 0, stream>>>(XAS, WPRJ, BPRJ, x, (float*)d_out);
}
